// Dynamics_45732811768369
// MI455X (gfx1250) — compile-verified, matrix operand order checked by reading, not run on hardware
//
#include <hip/hip_runtime.h>
#include <hip/hip_bf16.h>

#define NNODES 50000
#define NEDGES 800000
#define HDIM   128
#define INDIM  16
#define NLAYERS 2

typedef __attribute__((ext_vector_type(16))) __bf16 bf16x16;
typedef __attribute__((ext_vector_type(8)))  float  f32x8;

#if __has_builtin(__builtin_amdgcn_global_load_async_to_lds_b128) && \
    __has_builtin(__builtin_amdgcn_s_wait_asynccnt)
#define HAVE_ASYNC_LDS 1
typedef int v4i_t __attribute__((ext_vector_type(4)));
typedef __attribute__((address_space(1))) v4i_t* as1_v4i;
typedef __attribute__((address_space(3))) v4i_t* as3_v4i;
#else
#define HAVE_ASYNC_LDS 0
#endif

__device__ __forceinline__ float silu_f(float v) { return v / (1.0f + __expf(-v)); }

// WMMA 16x16x32 bf16 fragment loader.
// A-matrix layout (ISA 7.12.2): lane L holds row M = L&15; lanes 0-15 carry
// K = 0..7 and 16..23 in the 8 VGPRs, lanes 16-31 carry K = 8..15 and 24..31.
// B fragments use the same pattern on W stored transposed as [n][k] (lane's
// "row" is the output column). All strides chosen so p is 16B aligned.
__device__ __forceinline__ bf16x16 load_frag(const __bf16* base, int stride, int lane) {
  const __bf16* p = base + (size_t)(lane & 15) * stride + ((lane & 16) ? 8 : 0);
  union { int4 i[2]; bf16x16 v; } u;
  u.i[0] = *(const int4*)(p);
  u.i[1] = *(const int4*)(p + 16);
  return u.v;
}

#define WMMA_BF16(a, b, c) \
  __builtin_amdgcn_wmma_f32_16x16x32_bf16(false, (a), false, (b), (short)0, (c), false, false)

// ---------------------------------------------------------------- small kernels

__global__ __launch_bounds__(256) void transpose_bf16_kernel(
    const float* __restrict__ src, __bf16* __restrict__ dst, int K) {
  long t = (long)blockIdx.x * 256 + threadIdx.x;
  if (t >= (long)K * HDIM) return;
  int n = (int)(t / K);
  int k = (int)(t - (long)n * K);
  dst[t] = (__bf16)src[(size_t)k * HDIM + n];   // dst[n][k] = src[k][n]
}

__global__ __launch_bounds__(256) void radial_kernel(
    const float* __restrict__ x, const int* __restrict__ ei,
    float* __restrict__ radial, float* __restrict__ cdiff) {
  long e = (long)blockIdx.x * 256 + threadIdx.x;
  if (e >= NEDGES) return;
  int r = ei[e], c = ei[NEDGES + e];
  float dx = x[(size_t)r * 3 + 0] - x[(size_t)c * 3 + 0];
  float dy = x[(size_t)r * 3 + 1] - x[(size_t)c * 3 + 1];
  float dz = x[(size_t)r * 3 + 2] - x[(size_t)c * 3 + 2];
  float rad = dx * dx + dy * dy + dz * dz;
  radial[e] = rad;
  float inv = 1.0f / (sqrtf(rad + 1e-8f) + 1.0f);   // NORM_CONST = 1
  cdiff[e * 3 + 0] = dx * inv;
  cdiff[e * 3 + 1] = dy * inv;
  cdiff[e * 3 + 2] = dz * inv;
}

__global__ __launch_bounds__(256) void embed_kernel(
    const float* __restrict__ h, const float* __restrict__ W, const float* __restrict__ b,
    float* __restrict__ hf, __bf16* __restrict__ hbf) {
  long t = (long)blockIdx.x * 256 + threadIdx.x;
  if (t >= (long)NNODES * HDIM) return;
  int i = (int)(t >> 7), o = (int)(t & 127);
  float s = b[o];
#pragma unroll
  for (int k = 0; k < INDIM; ++k) s += h[(size_t)i * INDIM + k] * W[(size_t)k * HDIM + o];
  hf[t] = s;
  hbf[t] = (__bf16)s;
}

__global__ __launch_bounds__(256) void xupdate_kernel(
    float* __restrict__ xcur, const float* __restrict__ aggx,
    const float* __restrict__ rgmask, const float* __restrict__ nmask) {
  long i = (long)blockIdx.x * 256 + threadIdx.x;
  if (i >= NNODES) return;
  float rg = rgmask[i], nm = nmask[i];
#pragma unroll
  for (int d = 0; d < 3; ++d) {
    float v = (xcur[i * 3 + d] + aggx[i * 3 + d] * rg) * nm;   // aggx already /100
    xcur[i * 3 + d] = v;
  }
}

__global__ __launch_bounds__(256) void out_h_kernel(
    const float* __restrict__ hf, const float* __restrict__ W, const float* __restrict__ b,
    const float* __restrict__ nmask, float* __restrict__ out) {
  long t = (long)blockIdx.x * 256 + threadIdx.x;
  if (t >= (long)NNODES * INDIM) return;
  int i = (int)(t >> 4), o = (int)(t & 15);
  float s = b[o];
#pragma unroll 16
  for (int k = 0; k < HDIM; ++k) s += hf[(size_t)i * HDIM + k] * W[(size_t)k * INDIM + o];
  out[t] = s * nmask[i];
}

__global__ __launch_bounds__(256) void out_x_kernel(
    const float* __restrict__ xcur, float* __restrict__ out) {
  long t = (long)blockIdx.x * 256 + threadIdx.x;
  if (t >= (long)NNODES * 3) return;
  out[t] = xcur[t];
}

// ------------------------------------------------------------------ gather body
// Stage [h_row | h_col] (bf16, 256 features) for 32 edges into sIn.
// Async path: GLOBAL_LOAD_ASYNC_TO_LDS_B128 moves 16B chunks memory->LDS
// directly (no VGPR round-trip, tracked by ASYNCcnt). Invalid (tail) edges read
// node 0's row; their contribution is zeroed later via sMask, so no zero-fill
// or EXEC divergence is needed ahead of the WMMAs.
__device__ __forceinline__ void gather_edges(
    __bf16 (*sIn)[264], const __bf16* __restrict__ hbf,
    const int* __restrict__ ei, long e0, int tid) {
  for (int i = tid; i < 32 * 32; i += 256) {   // 16B chunks: 32 edges x 32 chunks
    int e = i >> 5, c = i & 31;
    long eg = e0 + e;
    bool v = eg < NEDGES;
    int src = v ? ((c < 16) ? ei[eg] : ei[NEDGES + eg]) : 0;
    int k8 = (c & 15) * 8;
    const __bf16* gp = hbf + (size_t)src * HDIM + k8;
    __bf16* lp = &sIn[e][(c < 16 ? 0 : 128) + k8];
#if HAVE_ASYNC_LDS
    __builtin_amdgcn_global_load_async_to_lds_b128(
        (as1_v4i)const_cast<__bf16*>(gp), (as3_v4i)lp, 0, 0);
#else
    *(int4*)lp = *(const int4*)gp;
#endif
  }
#if HAVE_ASYNC_LDS
  __builtin_amdgcn_s_wait_asynccnt(0);
#endif
}

// ---------------------------------------------------------------- edge MLP (WMMA)
// 32 edges per block; 8 waves: row-group = wave>>2 (16 edges), each wave covers
// 2 n-tiles of 16 output channels. GEMM1: [16,256]x[256,128] (+rank-1 for the 2
// scalar edge features). GEMM2: [16,128]x[128,128]. Scatter-add into agg.
__global__ __launch_bounds__(256) void edge_mlp_kernel(
    const __bf16* __restrict__ hbf, const int* __restrict__ ei,
    const float* __restrict__ radial, const float* __restrict__ dist0,
    const float* __restrict__ emask,
    const __bf16* __restrict__ W1t, const float* __restrict__ W1f,
    const float* __restrict__ b1,
    const __bf16* __restrict__ W2t, const float* __restrict__ b2,
    float* __restrict__ agg) {
  __shared__ __align__(16) __bf16 sIn[32][264];    // [h_row | h_col], padded stride
  __shared__ __align__(16) __bf16 sAct[32][136];
  __shared__ float sRad[32], sD0[32], sMask[32];
  __shared__ int sRow[32];

  const int tid = threadIdx.x;
  const long e0 = (long)blockIdx.x * 32;

  __builtin_prefetch(W1t, 0, 1);
  __builtin_prefetch(W2t, 0, 1);

  gather_edges(sIn, hbf, ei, e0, tid);
  if (tid < 32) {
    long eg = e0 + tid;
    bool v = eg < NEDGES;
    sRow[tid]  = v ? ei[eg] : 0;
    sRad[tid]  = v ? radial[eg] : 0.0f;
    sD0[tid]   = v ? dist0[eg] : 0.0f;
    sMask[tid] = v ? emask[eg] : 0.0f;             // 0 => contributes nothing
  }
  __syncthreads();

  const int lane = tid & 31, w = tid >> 5;
  const int rg = w >> 2, jb = (w & 3) * 2;
  const int nlo = lane & 15, mb = (lane & 16) ? 8 : 0;

  f32x8 acc0 = {}, acc1 = {};
  for (int kt = 0; kt < 8; ++kt) {
    bf16x16 a  = load_frag(&sIn[rg * 16][kt * 32], 264, lane);
    bf16x16 bA = load_frag(W1t + (size_t)(jb + 0) * 16 * 256 + kt * 32, 256, lane);
    bf16x16 bB = load_frag(W1t + (size_t)(jb + 1) * 16 * 256 + kt * 32, 256, lane);
    acc0 = WMMA_BF16(a, bA, acc0);
    acc1 = WMMA_BF16(a, bB, acc1);
  }
#pragma unroll
  for (int j = 0; j < 2; ++j) {
    f32x8 acc = j ? acc1 : acc0;
    int n = (jb + j) * 16 + nlo;
    float b1v  = b1[n];
    float w256 = W1f[256 * HDIM + n];              // rank-1: radial feature
    float w257 = W1f[257 * HDIM + n];              // rank-1: dist0 feature
#pragma unroll
    for (int r = 0; r < 8; ++r) {
      int m = rg * 16 + mb + r;
      float vv = acc[r] + b1v + sRad[m] * w256 + sD0[m] * w257;
      sAct[m][n] = (__bf16)silu_f(vv);
    }
  }
  __syncthreads();

  f32x8 c0 = {}, c1 = {};
  for (int kt = 0; kt < 4; ++kt) {
    bf16x16 a  = load_frag(&sAct[rg * 16][kt * 32], 136, lane);
    bf16x16 bA = load_frag(W2t + (size_t)(jb + 0) * 16 * 128 + kt * 32, 128, lane);
    bf16x16 bB = load_frag(W2t + (size_t)(jb + 1) * 16 * 128 + kt * 32, 128, lane);
    c0 = WMMA_BF16(a, bA, c0);
    c1 = WMMA_BF16(a, bB, c1);
  }
#pragma unroll
  for (int j = 0; j < 2; ++j) {
    f32x8 acc = j ? c1 : c0;
    int n = (jb + j) * 16 + nlo;
    float b2v = b2[n];
#pragma unroll
    for (int r = 0; r < 8; ++r) {
      int m = rg * 16 + mb + r;
      float vv = silu_f(acc[r] + b2v) * sMask[m] * 0.01f;   // fold /NORM_FACTOR
      atomicAdd(&agg[(size_t)sRow[m] * HDIM + n], vv);      // segment_sum by row
    }
  }
}

// ---------------------------------------------------------------- node MLP (WMMA)
__global__ __launch_bounds__(256) void node_mlp_kernel(
    __bf16* __restrict__ hbf, float* __restrict__ hf,
    const float* __restrict__ agg, const float* __restrict__ nmask,
    const __bf16* __restrict__ W1t, const float* __restrict__ b1,
    const __bf16* __restrict__ W2t, const float* __restrict__ b2) {
  __shared__ __align__(16) __bf16 sAgg[32][136];
  __shared__ __align__(16) __bf16 sAct[32][136];
  const int tid = threadIdx.x;
  const long n0 = (long)blockIdx.x * 32;

  __builtin_prefetch(W1t, 0, 1);
  __builtin_prefetch(W2t, 0, 1);

  for (int i = tid; i < 32 * HDIM; i += 256) {     // f32 -> bf16 conversion
    int e = i >> 7, c = i & 127;
    long node = n0 + e; if (node >= NNODES) node = NNODES - 1;
    sAgg[e][c] = (__bf16)agg[(size_t)node * HDIM + c];
  }
  __syncthreads();

  const int lane = tid & 31, w = tid >> 5;
  const int rg = w >> 2, jb = (w & 3) * 2;
  const int nlo = lane & 15, mb = (lane & 16) ? 8 : 0;

  f32x8 acc0 = {}, acc1 = {};
  for (int kt = 0; kt < 8; ++kt) {
    bf16x16 a;
    if (kt < 4) {
      // h half: rows are contiguous in hbf; tail-block overrun stays inside ws.
      a = load_frag(hbf + (size_t)(n0 + rg * 16) * HDIM + kt * 32, HDIM, lane);
    } else {
      a = load_frag(&sAgg[rg * 16][(kt - 4) * 32], 136, lane);
    }
    bf16x16 bA = load_frag(W1t + (size_t)(jb + 0) * 16 * 256 + kt * 32, 256, lane);
    bf16x16 bB = load_frag(W1t + (size_t)(jb + 1) * 16 * 256 + kt * 32, 256, lane);
    acc0 = WMMA_BF16(a, bA, acc0);
    acc1 = WMMA_BF16(a, bB, acc1);
  }
#pragma unroll
  for (int j = 0; j < 2; ++j) {
    f32x8 acc = j ? acc1 : acc0;
    int n = (jb + j) * 16 + nlo;
    float b1v = b1[n];
#pragma unroll
    for (int r = 0; r < 8; ++r) {
      int m = rg * 16 + mb + r;
      sAct[m][n] = (__bf16)silu_f(acc[r] + b1v);
    }
  }
  __syncthreads();

  f32x8 c0 = {}, c1 = {};
  for (int kt = 0; kt < 4; ++kt) {
    bf16x16 a  = load_frag(&sAct[rg * 16][kt * 32], 136, lane);
    bf16x16 bA = load_frag(W2t + (size_t)(jb + 0) * 16 * 128 + kt * 32, 128, lane);
    bf16x16 bB = load_frag(W2t + (size_t)(jb + 1) * 16 * 128 + kt * 32, 128, lane);
    c0 = WMMA_BF16(a, bA, c0);
    c1 = WMMA_BF16(a, bB, c1);
  }
#pragma unroll
  for (int j = 0; j < 2; ++j) {
    f32x8 acc = j ? c1 : c0;
    int n = (jb + j) * 16 + nlo;
    float b2v = b2[n];
#pragma unroll
    for (int r = 0; r < 8; ++r) {
      int m = rg * 16 + mb + r;
      long node = n0 + m;
      if (node < NNODES) {
        float vv = (hf[(size_t)node * HDIM + n] + acc[r] + b2v) * nmask[node];
        hf[(size_t)node * HDIM + n] = vv;
        hbf[(size_t)node * HDIM + n] = (__bf16)vv;
      }
    }
  }
}

// ---------------------------------------------------------------- coord MLP (WMMA)
__global__ __launch_bounds__(256) void coord_mlp_kernel(
    const __bf16* __restrict__ hbf, const int* __restrict__ ei,
    const float* __restrict__ radial, const float* __restrict__ dist0,
    const float* __restrict__ emask,
    const __bf16* __restrict__ W1t, const float* __restrict__ W1f,
    const float* __restrict__ b1,
    const __bf16* __restrict__ W2t, const float* __restrict__ b2,
    const float* __restrict__ W3,
    const float* __restrict__ cdiff, float* __restrict__ aggx) {
  __shared__ __align__(16) __bf16 sIn[32][264];
  __shared__ __align__(16) __bf16 sAct[32][136];
  __shared__ float sRad[32], sD0[32], sMask[32], sPhi[32];
  __shared__ int sRow[32];

  const int tid = threadIdx.x;
  const long e0 = (long)blockIdx.x * 32;

  __builtin_prefetch(W1t, 0, 1);
  __builtin_prefetch(W2t, 0, 1);

  gather_edges(sIn, hbf, ei, e0, tid);
  if (tid < 32) {
    long eg = e0 + tid;
    bool v = eg < NEDGES;
    sRow[tid]  = v ? ei[eg] : 0;
    sRad[tid]  = v ? radial[eg] : 0.0f;
    sD0[tid]   = v ? dist0[eg] : 0.0f;
    sMask[tid] = v ? emask[eg] : 0.0f;
    sPhi[tid]  = 0.0f;
  }
  __syncthreads();

  const int lane = tid & 31, w = tid >> 5;
  const int rg = w >> 2, jb = (w & 3) * 2;
  const int nlo = lane & 15, mb = (lane & 16) ? 8 : 0;

  f32x8 acc0 = {}, acc1 = {};
  for (int kt = 0; kt < 8; ++kt) {
    bf16x16 a  = load_frag(&sIn[rg * 16][kt * 32], 264, lane);
    bf16x16 bA = load_frag(W1t + (size_t)(jb + 0) * 16 * 256 + kt * 32, 256, lane);
    bf16x16 bB = load_frag(W1t + (size_t)(jb + 1) * 16 * 256 + kt * 32, 256, lane);
    acc0 = WMMA_BF16(a, bA, acc0);
    acc1 = WMMA_BF16(a, bB, acc1);
  }
#pragma unroll
  for (int j = 0; j < 2; ++j) {
    f32x8 acc = j ? acc1 : acc0;
    int n = (jb + j) * 16 + nlo;
    float b1v  = b1[n];
    float w256 = W1f[256 * HDIM + n];
    float w257 = W1f[257 * HDIM + n];
#pragma unroll
    for (int r = 0; r < 8; ++r) {
      int m = rg * 16 + mb + r;
      float vv = acc[r] + b1v + sRad[m] * w256 + sD0[m] * w257;
      sAct[m][n] = (__bf16)silu_f(vv);
    }
  }
  __syncthreads();

  f32x8 c0 = {}, c1 = {};
  for (int kt = 0; kt < 4; ++kt) {
    bf16x16 a  = load_frag(&sAct[rg * 16][kt * 32], 136, lane);
    bf16x16 bA = load_frag(W2t + (size_t)(jb + 0) * 16 * 128 + kt * 32, 128, lane);
    bf16x16 bB = load_frag(W2t + (size_t)(jb + 1) * 16 * 128 + kt * 32, 128, lane);
    c0 = WMMA_BF16(a, bA, c0);
    c1 = WMMA_BF16(a, bB, c1);
  }
  // phi[m] = sum_n silu(act2[m][n] + b2[n]) * W3[n]  -> LDS f32 atomics
  float part[8] = {0.f, 0.f, 0.f, 0.f, 0.f, 0.f, 0.f, 0.f};
#pragma unroll
  for (int j = 0; j < 2; ++j) {
    f32x8 acc = j ? c1 : c0;
    int n = (jb + j) * 16 + nlo;
    float b2v = b2[n], w3 = W3[n];
#pragma unroll
    for (int r = 0; r < 8; ++r) part[r] += silu_f(acc[r] + b2v) * w3;
  }
#pragma unroll
  for (int r = 0; r < 8; ++r) atomicAdd(&sPhi[rg * 16 + mb + r], part[r]);
  __syncthreads();

  if (tid < 32) {
    long eg = e0 + tid;
    if (eg < NEDGES) {
      float phi = sPhi[tid] * sMask[tid] * 0.01f;    // fold /NORM_FACTOR
      int dstn = sRow[tid];
#pragma unroll
      for (int d = 0; d < 3; ++d)
        atomicAdd(&aggx[(size_t)dstn * 3 + d], cdiff[eg * 3 + d] * phi);
    }
  }
}

// ---------------------------------------------------------------- launch

extern "C" void kernel_launch(void* const* d_in, const int* in_sizes, int n_in,
                              void* d_out, int out_size, void* d_ws, size_t ws_size,
                              hipStream_t stream) {
  const float* h_in   = (const float*)d_in[0];
  const float* x_in   = (const float*)d_in[1];
  const int*   ei     = (const int*)d_in[2];
  const float* nmask  = (const float*)d_in[3];
  const float* rgmask = (const float*)d_in[4];
  const float* emask  = (const float*)d_in[5];
  const float* embW   = (const float*)d_in[6];
  const float* embB   = (const float*)d_in[7];
  const float* eoW    = (const float*)d_in[8];
  const float* eoB    = (const float*)d_in[9];
  const float* eW1    = (const float*)d_in[10];
  const float* eB1    = (const float*)d_in[11];
  const float* eW2    = (const float*)d_in[12];
  const float* eB2    = (const float*)d_in[13];
  const float* nW1    = (const float*)d_in[14];
  const float* nB1    = (const float*)d_in[15];
  const float* nW2    = (const float*)d_in[16];
  const float* nB2    = (const float*)d_in[17];
  const float* cW1    = (const float*)d_in[18];
  const float* cB1    = (const float*)d_in[19];
  const float* cW2    = (const float*)d_in[20];
  const float* cB2    = (const float*)d_in[21];
  const float* cW3    = (const float*)d_in[22];

  char* base = (char*)d_ws;
  size_t off = 0;
  auto alloc = [&](size_t bytes) -> void* {
    void* p = base + off;
    off = (off + bytes + 255) & ~(size_t)255;
    return p;
  };
  __bf16* hbf   = (__bf16*)alloc((size_t)NNODES * HDIM * 2);
  float*  hf    = (float*) alloc((size_t)NNODES * HDIM * 4);
  float*  agg   = (float*) alloc((size_t)NNODES * HDIM * 4);
  float*  aggx  = (float*) alloc((size_t)NNODES * 3 * 4);
  float*  xcur  = (float*) alloc((size_t)NNODES * 3 * 4);
  float*  dist0 = (float*) alloc((size_t)NEDGES * 4);
  float*  radial= (float*) alloc((size_t)NEDGES * 4);
  float*  cdiff = (float*) alloc((size_t)NEDGES * 3 * 4);
  __bf16* eW1t[NLAYERS]; __bf16* eW2t[NLAYERS];
  __bf16* nW1t[NLAYERS]; __bf16* nW2t[NLAYERS];
  __bf16* cW1t[NLAYERS]; __bf16* cW2t[NLAYERS];
  for (int l = 0; l < NLAYERS; ++l) {
    eW1t[l] = (__bf16*)alloc((size_t)HDIM * 256 * 2);
    eW2t[l] = (__bf16*)alloc((size_t)HDIM * 128 * 2);
    nW1t[l] = (__bf16*)alloc((size_t)HDIM * 256 * 2);
    nW2t[l] = (__bf16*)alloc((size_t)HDIM * 128 * 2);
    cW1t[l] = (__bf16*)alloc((size_t)HDIM * 256 * 2);
    cW2t[l] = (__bf16*)alloc((size_t)HDIM * 128 * 2);
  }

  auto T = [&](const float* src, __bf16* dst, int K) {
    long tot = (long)K * HDIM;
    transpose_bf16_kernel<<<(unsigned)((tot + 255) / 256), 256, 0, stream>>>(src, dst, K);
  };
  for (int l = 0; l < NLAYERS; ++l) {
    T(eW1 + (size_t)l * 258 * HDIM, eW1t[l], 256);
    T(eW2 + (size_t)l * 128 * HDIM, eW2t[l], 128);
    T(nW1 + (size_t)l * 256 * HDIM, nW1t[l], 256);
    T(nW2 + (size_t)l * 128 * HDIM, nW2t[l], 128);
    T(cW1 + (size_t)l * 258 * HDIM, cW1t[l], 256);
    T(cW2 + (size_t)l * 128 * HDIM, cW2t[l], 128);
  }

  (void)hipMemcpyAsync(xcur, x_in, (size_t)NNODES * 3 * 4, hipMemcpyDeviceToDevice, stream);

  // EGNN-level distances (from initial x); cdiff output overwritten per layer.
  radial_kernel<<<(NEDGES + 255) / 256, 256, 0, stream>>>(x_in, ei, dist0, cdiff);

  embed_kernel<<<(unsigned)(((long)NNODES * HDIM + 255) / 256), 256, 0, stream>>>(
      h_in, embW, embB, hf, hbf);

  const unsigned eblocks = (NEDGES + 31) / 32;
  const unsigned nblocks = (NNODES + 31) / 32;
  for (int l = 0; l < NLAYERS; ++l) {
    radial_kernel<<<(NEDGES + 255) / 256, 256, 0, stream>>>(xcur, ei, radial, cdiff);
    (void)hipMemsetAsync(agg,  0, (size_t)NNODES * HDIM * 4, stream);
    (void)hipMemsetAsync(aggx, 0, (size_t)NNODES * 3 * 4, stream);

    edge_mlp_kernel<<<eblocks, 256, 0, stream>>>(
        hbf, ei, radial, dist0, emask,
        eW1t[l], eW1 + (size_t)l * 258 * HDIM, eB1 + (size_t)l * HDIM,
        eW2t[l], eB2 + (size_t)l * HDIM, agg);

    node_mlp_kernel<<<nblocks, 256, 0, stream>>>(
        hbf, hf, agg, nmask,
        nW1t[l], nB1 + (size_t)l * HDIM,
        nW2t[l], nB2 + (size_t)l * HDIM);

    coord_mlp_kernel<<<eblocks, 256, 0, stream>>>(
        hbf, ei, radial, dist0, emask,
        cW1t[l], cW1 + (size_t)l * 258 * HDIM, cB1 + (size_t)l * HDIM,
        cW2t[l], cB2 + (size_t)l * HDIM, cW3 + (size_t)l * HDIM,
        cdiff, aggx);

    xupdate_kernel<<<(NNODES + 255) / 256, 256, 0, stream>>>(xcur, aggx, rgmask, nmask);
  }

  out_h_kernel<<<(unsigned)(((long)NNODES * INDIM + 255) / 256), 256, 0, stream>>>(
      hf, eoW, eoB, nmask, (float*)d_out);
  out_x_kernel<<<(unsigned)(((long)NNODES * 3 + 255) / 256), 256, 0, stream>>>(
      xcur, (float*)d_out + (size_t)NNODES * INDIM);
}
